// FasterRCNN_65841848648203
// MI455X (gfx1250) — matrix operand order repaired and checked by reading, not compiled, not run on hardware
//
#include <hip/hip_runtime.h>
#include <hip/hip_bf16.h>
#include <math.h>
#include <stdint.h>

typedef __attribute__((ext_vector_type(16))) _Float16 v16h;
typedef __attribute__((ext_vector_type(8)))  float    v8f;

#define IMGSZ 608
#define FHW 38
#define NANCH 12996          // 38*38*9
#define PRE_NMS 1000
#define POST_NMS 512
#define NCLS 91
#define DETS 100
#define NEGV (-1.0e30f)
#define CLIP_DWH 4.135166556742356f   // log(1000/16)

// async global->LDS path (gfx1250); falls back to sync staging if builtins absent
#if __has_builtin(__builtin_amdgcn_global_load_async_to_lds_b128) && __has_builtin(__builtin_amdgcn_s_wait_asynccnt)
#define USE_ASYNC 1
#else
#define USE_ASYNC 0
#endif
// builtin takes (vec4i AS1* global_src, vec4i AS3* lds_dst, imm offset, imm cpol)
typedef int vint4_t __attribute__((__vector_size__(4 * sizeof(int))));
#define GASP(p) ((__attribute__((address_space(1))) vint4_t*)(uintptr_t)(p))
#define LASP(p) ((__attribute__((address_space(3))) vint4_t*)(unsigned int)(uintptr_t)(p))

// ---------------------------------------------------------------- normalize
__global__ void k_normalize(const float* __restrict__ img, float* __restrict__ out, int n) {
  int i = blockIdx.x * blockDim.x + threadIdx.x;
  if (i >= n) return;
  int c = (i / (IMGSZ * IMGSZ)) % 3;
  const float mean[3] = {0.485f, 0.456f, 0.406f};
  const float stdv[3] = {0.229f, 0.224f, 0.225f};
  out[i] = (img[i] - mean[c]) / stdv[c];
}

// ---------------------------------------------------------------- generic conv (NCHW, OIHW)
__global__ void k_conv(const float* __restrict__ x, const float* __restrict__ w,
                       const float* __restrict__ bias, float* __restrict__ y,
                       int Cin, int Hin, int Win, int Cout, int Hout, int Wout,
                       int KS, int S, int P, int relu, int n) {
  int i = blockIdx.x * blockDim.x + threadIdx.x;
  if (i >= n) return;
  int ox = i % Wout; int t = i / Wout;
  int oy = t % Hout; t /= Hout;
  int oc = t % Cout; int b = t / Cout;
  float acc = bias[oc];
  int iy0 = oy * S - P, ix0 = ox * S - P;
  const float* wp = w + (size_t)oc * Cin * KS * KS;
  for (int ic = 0; ic < Cin; ++ic) {
    const float* xp = x + ((size_t)(b * Cin + ic) * Hin) * Win;
    const float* wc = wp + (size_t)ic * KS * KS;
    for (int kh = 0; kh < KS; ++kh) {
      int iy = iy0 + kh;
      if (iy < 0 || iy >= Hin) continue;
      for (int kw = 0; kw < KS; ++kw) {
        int ix = ix0 + kw;
        if (ix < 0 || ix >= Win) continue;
        acc += xp[iy * Win + ix] * wc[kh * KS + kw];
      }
    }
  }
  y[i] = relu ? fmaxf(acc, 0.f) : acc;
}

// ---------------------------------------------------------------- proposal decode + sigmoid
__global__ void k_props(const float* __restrict__ obj, const float* __restrict__ reg,
                        float* __restrict__ props, float* __restrict__ scores, int n) {
  int i = blockIdx.x * blockDim.x + threadIdx.x;
  if (i >= n) return;                       // n = 2*NANCH
  int b = i / NANCH, ai = i % NANCH;
  int cell = ai / 9, a = ai % 9;
  int gy = cell / FHW, gx = cell % FHW;
  int r = a / 3, s = a % 3;
  const float sizes[3]  = {128.f, 256.f, 512.f};
  const float ratios[3] = {0.5f, 1.f, 2.f};
  float hr = sqrtf(ratios[r]); float wr = 1.f / hr;
  float wsz = wr * sizes[s], hsz = hr * sizes[s];
  float acx = gx * 16.f, acy = gy * 16.f;   // anchor center = shift (box centered)
  float aw = wsz, ah = hsz;
  float d0 = reg[((size_t)(b * 36 + a * 4 + 0) * FHW + gy) * FHW + gx];
  float d1 = reg[((size_t)(b * 36 + a * 4 + 1) * FHW + gy) * FHW + gx];
  float d2 = reg[((size_t)(b * 36 + a * 4 + 2) * FHW + gy) * FHW + gx];
  float d3 = reg[((size_t)(b * 36 + a * 4 + 3) * FHW + gy) * FHW + gx];
  float dw = fminf(d2, CLIP_DWH), dh = fminf(d3, CLIP_DWH);
  float px = d0 * aw + acx, py = d1 * ah + acy;
  float pw = expf(dw) * aw, ph = expf(dh) * ah;
  props[(size_t)i * 4 + 0] = fminf(fmaxf(px - 0.5f * pw, 0.f), 608.f);
  props[(size_t)i * 4 + 1] = fminf(fmaxf(py - 0.5f * ph, 0.f), 608.f);
  props[(size_t)i * 4 + 2] = fminf(fmaxf(px + 0.5f * pw, 0.f), 608.f);
  props[(size_t)i * 4 + 3] = fminf(fmaxf(py + 0.5f * ph, 0.f), 608.f);
  float o = obj[((size_t)(b * 9 + a) * FHW + gy) * FHW + gx];
  scores[i] = 1.f / (1.f + expf(-o));
}

// ---------------------------------------------------------------- top-k 1000 (per batch)
__global__ __launch_bounds__(256) void k_topk_pre(const float* __restrict__ scores,
                                                  const float4* __restrict__ props,
                                                  float* __restrict__ top_s,
                                                  float4* __restrict__ top_b) {
  int b = blockIdx.x, tid = threadIdx.x;
  __shared__ float s[NANCH];
  __shared__ float rv[256];
  __shared__ int ri[256];
  for (int i = tid; i < NANCH; i += 256) s[i] = scores[b * NANCH + i];
  __syncthreads();
  for (int t = 0; t < PRE_NMS; ++t) {
    float bv = -3.0e38f; int bi = NANCH;
    for (int i = tid; i < NANCH; i += 256)
      if (s[i] > bv) { bv = s[i]; bi = i; }
    rv[tid] = bv; ri[tid] = bi;
    __syncthreads();
    for (int off = 128; off > 0; off >>= 1) {
      if (tid < off) {
        if (rv[tid + off] > rv[tid] || (rv[tid + off] == rv[tid] && ri[tid + off] < ri[tid])) {
          rv[tid] = rv[tid + off]; ri[tid] = ri[tid + off];
        }
      }
      __syncthreads();
    }
    if (tid == 0) {
      int j = ri[0];
      top_s[b * PRE_NMS + t] = rv[0];
      top_b[b * PRE_NMS + t] = props[b * NANCH + j];
      s[j] = -3.0e38f;
    }
    __syncthreads();
  }
}

// ---------------------------------------------------------------- RPN NMS -> 512 proposals
__global__ __launch_bounds__(256) void k_rpn_nms(const float* __restrict__ top_s,
                                                 const float4* __restrict__ top_b,
                                                 float4* __restrict__ pboxes,
                                                 int* __restrict__ valid) {
  int b = blockIdx.x, tid = threadIdx.x;
  __shared__ float4 bx[PRE_NMS];
  __shared__ float sc[PRE_NMS];
  __shared__ float ar[PRE_NMS];
  __shared__ float rv[256];
  __shared__ int ri[256];
  __shared__ float4 cur;
  __shared__ int curj;
  for (int i = tid; i < PRE_NMS; i += 256) {
    float4 bb = top_b[b * PRE_NMS + i];
    bx[i] = bb;
    sc[i] = top_s[b * PRE_NMS + i];
    ar[i] = (bb.z - bb.x) * (bb.w - bb.y);
  }
  __syncthreads();
  for (int t = 0; t < POST_NMS; ++t) {
    float bv = -3.0e38f; int bi = PRE_NMS;
    for (int i = tid; i < PRE_NMS; i += 256)
      if (sc[i] > bv) { bv = sc[i]; bi = i; }
    rv[tid] = bv; ri[tid] = bi;
    __syncthreads();
    for (int off = 128; off > 0; off >>= 1) {
      if (tid < off) {
        if (rv[tid + off] > rv[tid] || (rv[tid + off] == rv[tid] && ri[tid + off] < ri[tid])) {
          rv[tid] = rv[tid + off]; ri[tid] = ri[tid + off];
        }
      }
      __syncthreads();
    }
    if (tid == 0) {
      int j = ri[0]; float sv = rv[0];
      curj = j; cur = bx[j];
      int ok = sv > (NEGV * 0.5f);
      valid[b * POST_NMS + t] = ok;
      pboxes[b * POST_NMS + t] = ok ? bx[j] : make_float4(0.f, 0.f, 16.f, 16.f);
    }
    __syncthreads();
    float4 bj = cur; int j = curj;
    float aj = (bj.z - bj.x) * (bj.w - bj.y);
    for (int i = tid; i < PRE_NMS; i += 256) {
      float iw = fmaxf(fminf(bj.z, bx[i].z) - fmaxf(bj.x, bx[i].x), 0.f);
      float ih = fmaxf(fminf(bj.w, bx[i].w) - fmaxf(bj.y, bx[i].y), 0.f);
      float inter = iw * ih;
      float iou = inter / (aj + ar[i] - inter + 1e-9f);
      if (iou > 0.7f || i == j) sc[i] = NEGV;
    }
    __syncthreads();
  }
}

// ---------------------------------------------------------------- ROI align (f16 out for WMMA)
__global__ void k_roi_align(const float* __restrict__ f3, const float4* __restrict__ pboxes,
                            _Float16* __restrict__ out, int n) {
  int i = blockIdx.x * blockDim.x + threadIdx.x;
  if (i >= n) return;                        // n = 1024*12544
  int k = i % 12544; int row = i / 12544;
  int c = k / 49; int pp = k % 49; int py = pp / 7; int px = pp % 7;
  int b = row >> 9;
  float4 roi = pboxes[row];
  float x1 = roi.x * 0.0625f, y1 = roi.y * 0.0625f;
  float x2 = roi.z * 0.0625f, y2 = roi.w * 0.0625f;
  float bw = fmaxf(x2 - x1, 1.f) * (1.f / 7.f);
  float bh = fmaxf(y2 - y1, 1.f) * (1.f / 7.f);
  const float* fp = f3 + (size_t)(b * 256 + c) * (FHW * FHW);
  float acc = 0.f;
  for (int sy = 0; sy < 2; ++sy) {
    float gy = (py * 2 + sy + 0.5f) * 0.5f;
    float ys = fminf(fmaxf(y1 + gy * bh, 0.f), 37.f);
    int y0 = (int)floorf(ys); int y1i = min(y0 + 1, 37);
    float ly = ys - (float)y0, hy = 1.f - ly;
    for (int sx = 0; sx < 2; ++sx) {
      float gx = (px * 2 + sx + 0.5f) * 0.5f;
      float xs = fminf(fmaxf(x1 + gx * bw, 0.f), 37.f);
      int x0 = (int)floorf(xs); int x1i = min(x0 + 1, 37);
      float lx = xs - (float)x0, hx = 1.f - lx;
      acc += fp[y0 * FHW + x0] * hy * hx + fp[y0 * FHW + x1i] * hy * lx +
             fp[y1i * FHW + x0] * ly * hx + fp[y1i * FHW + x1i] * ly * lx;
    }
  }
  out[(size_t)i] = (_Float16)(acc * 0.25f);
}

// ---------------------------------------------------------------- f32 -> f16 convert
__global__ void k_cvt(const float* __restrict__ src, _Float16* __restrict__ dst, int n) {
  int i = blockIdx.x * blockDim.x + threadIdx.x;
  if (i >= n) return;
  dst[i] = (_Float16)src[i];
}

// ---------------------------------------------------------------- WMMA GEMM: C = relu(A*B + bias)
// A: MxK f16 row-major, B: KxN f16 row-major. M%128==0, N%64==0, K%32==0.
// Double-buffered LDS. A staged via async global->LDS B128 (gfx1250); B staged with an
// in-flight transpose so both A and B fragments are contiguous 16B ds_load_b128 reads.
#define MBLK 128
#define NBLK 64
#define KBLK 32
#define KPAD 40   // 80B row stride (16B aligned)
__global__ __launch_bounds__(256) void k_gemm_wmma(const _Float16* __restrict__ A,
                                                   const _Float16* __restrict__ Bm,
                                                   const float* __restrict__ bias,
                                                   float* Cf, _Float16* Ch,
                                                   int M, int N, int K, int relu) {
  __shared__ _Float16 sA[2][MBLK][KPAD];     // [m][k], padded rows
  __shared__ _Float16 sBT[2][NBLK][KPAD];    // [n][k], transposed, padded rows
  int tid = threadIdx.x;
  int lane = tid & 31, wave = tid >> 5;
  int m0 = blockIdx.y * MBLK, n0 = blockIdx.x * NBLK;
  v8f acc[4] = {};
  int mw = wave * 16;
  int arow = mw + (lane & 15);
  int kg = (lane >> 4) * 8;            // A fragment K group
  int kbB = (lane & 16) ? 16 : 0;      // B fragment K base
  int ncl = lane & 15;                 // B fragment column within 16-wide tile

  // per-thread staging coordinates (16-byte aligned everywhere)
  int mA = tid >> 1;                   // 0..127
  int kA = (tid & 1) * 16;             // 0 or 16
  int kB = tid >> 3;                   // 0..31
  int nB = (tid & 7) * 8;              // 0..56 step 8
  const _Float16* gA = A + (size_t)(m0 + mA) * K + kA;
  const _Float16* gB = Bm + (size_t)kB * N + n0 + nB;

  auto stage = [&](int it, int buf) {
    const _Float16* ga = gA + (size_t)it * KBLK;
    const _Float16* gb = gB + (size_t)it * KBLK * N;
    _Float16* la = &sA[buf][mA][kA];
#if USE_ASYNC
    __builtin_amdgcn_global_load_async_to_lds_b128(GASP(ga),     LASP(la),     0, 0);
    __builtin_amdgcn_global_load_async_to_lds_b128(GASP(ga + 8), LASP(la + 8), 0, 0);
#else
    ((uint4*)la)[0] = ((const uint4*)ga)[0];
    ((uint4*)la)[1] = ((const uint4*)ga)[1];
#endif
    // B: load 8 contiguous halves of row kB, scatter transposed into sBT[n][k]
    alignas(16) _Float16 tb[8];
    *(uint4*)tb = ((const uint4*)gb)[0];
    #pragma unroll
    for (int j = 0; j < 8; ++j) sBT[buf][nB + j][kB] = tb[j];
  };

  int nk = K / KBLK;
  stage(0, 0);
  for (int it = 0; it < nk; ++it) {
    int buf = it & 1;
#if USE_ASYNC
    __builtin_amdgcn_s_wait_asynccnt(0);
#endif
    __syncthreads();
    if (it + 1 < nk) stage(it + 1, buf ^ 1);

    union { v16h v; _Float16 e[16]; } af;
    #pragma unroll
    for (int h = 0; h < 16; ++h) {
      int kk = 2 * kg + h;
      af.e[h] = sA[buf][arow][kk];
    }
    #pragma unroll
    for (int nt = 0; nt < 4; ++nt) {
      union { v16h v; _Float16 e[16]; } bf;
      int ncol = nt * 16 + ncl;
      #pragma unroll
      for (int h = 0; h < 16; ++h) bf.e[h] = sBT[buf][ncol][kbB + h];
      acc[nt] = __builtin_amdgcn_wmma_f32_16x16x32_f16(false, af.v, false, bf.v,
                                                       (short)0, acc[nt], false, false);
    }
  }
  int rsel = (lane & 16) ? 8 : 0;
  #pragma unroll
  for (int nt = 0; nt < 4; ++nt) {
    union { v8f v; float e[8]; } u; u.v = acc[nt];
    int ncol = n0 + nt * 16 + ncl;
    float bv = bias[ncol];
    #pragma unroll
    for (int vv = 0; vv < 8; ++vv) {
      int cidx = (m0 + mw + rsel + vv) * N + ncol;   // fits in 32 bits
      float val = u.e[vv] + bv;
      if (relu) val = fmaxf(val, 0.f);
      if (Cf) Cf[cidx] = val;
      if (Ch) Ch[cidx] = (_Float16)val;
    }
  }
}

// ---------------------------------------------------------------- small FC heads (f32)
__global__ void k_fc(const float* __restrict__ X, const float* __restrict__ W,
                     const float* __restrict__ bias, float* __restrict__ Y,
                     int Kdim, int Ndim, int n) {
  int i = blockIdx.x * blockDim.x + threadIdx.x;
  if (i >= n) return;
  int c = i % Ndim, row = i / Ndim;
  float acc = bias[c];
  const float* xr = X + (size_t)row * Kdim;
  for (int k = 0; k < Kdim; ++k) acc += xr[k] * W[(size_t)k * Ndim + c];
  Y[i] = acc;
}

// ---------------------------------------------------------------- softmax + box decode -> per-class arrays
__global__ void k_post(const float* __restrict__ cls_out, const float* __restrict__ reg_out,
                       const float4* __restrict__ pboxes, const int* __restrict__ valid,
                       float4* __restrict__ cb, float* __restrict__ cs, int n) {
  int row = blockIdx.x * blockDim.x + threadIdx.x;
  if (row >= n) return;                       // n = 1024
  int b = row >> 9, r = row & 511;
  const float* lg = cls_out + (size_t)row * NCLS;
  float mx = lg[0];
  for (int c = 1; c < NCLS; ++c) mx = fmaxf(mx, lg[c]);
  float sum = 0.f;
  for (int c = 0; c < NCLS; ++c) sum += expf(lg[c] - mx);
  float inv = 1.f / sum;
  float4 pb = pboxes[row];
  float w = pb.z - pb.x, h = pb.w - pb.y;
  float cx = pb.x + 0.5f * w, cy = pb.y + 0.5f * h;
  int ok = valid[row];
  for (int c = 1; c < NCLS; ++c) {
    float p = expf(lg[c] - mx) * inv;
    const float* dt = reg_out + (size_t)row * (NCLS * 4) + c * 4;
    float dx = dt[0] * 0.1f, dy = dt[1] * 0.1f;
    float dw = fminf(dt[2] * 0.2f, CLIP_DWH), dh = fminf(dt[3] * 0.2f, CLIP_DWH);
    float px = dx * w + cx, py = dy * h + cy;
    float pw = expf(dw) * w, ph = expf(dh) * h;
    float bx1 = fminf(fmaxf(px - 0.5f * pw, 0.f), 608.f);
    float by1 = fminf(fmaxf(py - 0.5f * ph, 0.f), 608.f);
    float bx2 = fminf(fmaxf(px + 0.5f * pw, 0.f), 608.f);
    float by2 = fminf(fmaxf(py + 0.5f * ph, 0.f), 608.f);
    int o = (b * 90 + (c - 1)) * POST_NMS + r;
    cb[o] = make_float4(bx1, by1, bx2, by2);
    cs[o] = (p > 0.05f && ok) ? p : NEGV;
  }
}

// ---------------------------------------------------------------- per-class NMS (180 blocks)
__global__ __launch_bounds__(256) void k_class_nms(const float4* __restrict__ cb,
                                                   const float* __restrict__ cs,
                                                   float4* __restrict__ kb,
                                                   float* __restrict__ sc2) {
  int blk = blockIdx.x, tid = threadIdx.x;     // blk = b*90 + class_index
  __shared__ float4 bx[POST_NMS];
  __shared__ float sc[POST_NMS];
  __shared__ float ar[POST_NMS];
  __shared__ float rv[256];
  __shared__ int ri[256];
  __shared__ float4 cur;
  __shared__ int curj;
  for (int i = tid; i < POST_NMS; i += 256) {
    float4 bb = cb[blk * POST_NMS + i];
    bx[i] = bb;
    sc[i] = cs[blk * POST_NMS + i];
    ar[i] = (bb.z - bb.x) * (bb.w - bb.y);
  }
  __syncthreads();
  for (int t = 0; t < DETS; ++t) {
    float bv = -3.0e38f; int bi = POST_NMS;
    for (int i = tid; i < POST_NMS; i += 256)
      if (sc[i] > bv) { bv = sc[i]; bi = i; }
    rv[tid] = bv; ri[tid] = bi;
    __syncthreads();
    for (int off = 128; off > 0; off >>= 1) {
      if (tid < off) {
        if (rv[tid + off] > rv[tid] || (rv[tid + off] == rv[tid] && ri[tid + off] < ri[tid])) {
          rv[tid] = rv[tid + off]; ri[tid] = ri[tid + off];
        }
      }
      __syncthreads();
    }
    if (tid == 0) {
      int j = ri[0];
      curj = j; cur = bx[j];
      kb[blk * DETS + t] = bx[j];
      sc2[blk * DETS + t] = rv[0];
    }
    __syncthreads();
    float4 bj = cur; int j = curj;
    float aj = (bj.z - bj.x) * (bj.w - bj.y);
    for (int i = tid; i < POST_NMS; i += 256) {
      float iw = fmaxf(fminf(bj.z, bx[i].z) - fmaxf(bj.x, bx[i].x), 0.f);
      float ih = fmaxf(fminf(bj.w, bx[i].w) - fmaxf(bj.y, bx[i].y), 0.f);
      float inter = iw * ih;
      float iou = inter / (aj + ar[i] - inter + 1e-9f);
      if (iou > 0.5f || i == j) sc[i] = NEGV;
    }
    __syncthreads();
  }
}

// ---------------------------------------------------------------- final top-100 over 9000 + outputs
__global__ __launch_bounds__(256) void k_final(const float* __restrict__ sc2,
                                               const float4* __restrict__ kb,
                                               float* __restrict__ out) {
  int b = blockIdx.x, tid = threadIdx.x;
  __shared__ float s[9000];
  __shared__ float rv[256];
  __shared__ int ri[256];
  for (int i = tid; i < 9000; i += 256) s[i] = sc2[b * 9000 + i];
  __syncthreads();
  for (int t = 0; t < DETS; ++t) {
    float bv = -3.0e38f; int bi = 9000;
    for (int i = tid; i < 9000; i += 256)
      if (s[i] > bv) { bv = s[i]; bi = i; }
    rv[tid] = bv; ri[tid] = bi;
    __syncthreads();
    for (int off = 128; off > 0; off >>= 1) {
      if (tid < off) {
        if (rv[tid + off] > rv[tid] || (rv[tid + off] == rv[tid] && ri[tid + off] < ri[tid])) {
          rv[tid] = rv[tid + off]; ri[tid] = ri[tid + off];
        }
      }
      __syncthreads();
    }
    if (tid == 0) {
      int j = ri[0]; float v = rv[0];
      int ok = v > (NEGV * 0.5f);
      float4 bb = kb[b * 9000 + j];
      int oi = b * DETS + t;
      out[oi * 4 + 0] = ok ? bb.x : 0.f;
      out[oi * 4 + 1] = ok ? bb.y : 0.f;
      out[oi * 4 + 2] = ok ? bb.z : 0.f;
      out[oi * 4 + 3] = ok ? bb.w : 0.f;
      out[800 + oi]  = ok ? v : 0.f;
      out[1000 + oi] = ok ? (float)(j / DETS + 1) : 0.f;
      s[j] = -3.0e38f;
    }
    __syncthreads();
  }
}

// ================================================================ host
extern "C" void kernel_launch(void* const* d_in, const int* in_sizes, int n_in,
                              void* d_out, int out_size, void* d_ws, size_t ws_size,
                              hipStream_t stream) {
  (void)in_sizes; (void)n_in; (void)out_size; (void)ws_size;
  const float* images = (const float*)d_in[0];
  const float* bb_w1 = (const float*)d_in[1];  const float* bb_b1 = (const float*)d_in[2];
  const float* bb_w2 = (const float*)d_in[3];  const float* bb_b2 = (const float*)d_in[4];
  const float* bb_w3 = (const float*)d_in[5];  const float* bb_b3 = (const float*)d_in[6];
  const float* rpn_cw = (const float*)d_in[7]; const float* rpn_cb = (const float*)d_in[8];
  const float* rpn_sw = (const float*)d_in[9]; const float* rpn_sb = (const float*)d_in[10];
  const float* rpn_rw = (const float*)d_in[11];const float* rpn_rb = (const float*)d_in[12];
  const float* fc1_w = (const float*)d_in[13]; const float* fc1_b = (const float*)d_in[14];
  const float* fc2_w = (const float*)d_in[15]; const float* fc2_b = (const float*)d_in[16];
  const float* cls_w = (const float*)d_in[17]; const float* cls_b = (const float*)d_in[18];
  const float* reg_w = (const float*)d_in[19]; const float* reg_b = (const float*)d_in[20];

  char* base = (char*)d_ws; size_t off = 0;
  auto carve = [&](size_t bytes) -> void* {
    void* p = base + off;
    off += (bytes + 255) & ~(size_t)255;
    return p;
  };
  float* xn      = (float*)carve((size_t)2217984 * 4);   // 2*3*608*608
  float* f1      = (float*)carve((size_t)2957312 * 4);   // 2*64*152*152
  float* f2      = (float*)carve((size_t)1478656 * 4);   // 2*128*76*76
  float* f3      = (float*)carve((size_t)739328 * 4);    // 2*256*38*38
  float* trpn    = (float*)carve((size_t)739328 * 4);
  float* objb    = (float*)carve((size_t)25992 * 4);
  float* regb    = (float*)carve((size_t)103968 * 4);
  float* props   = (float*)carve((size_t)103968 * 4);
  float* scores  = (float*)carve((size_t)25992 * 4);
  float* top_s   = (float*)carve((size_t)2000 * 4);
  float4* top_b  = (float4*)carve((size_t)2000 * 16);
  float4* pboxes = (float4*)carve((size_t)1024 * 16);
  int* validb    = (int*)carve((size_t)1024 * 4);
  _Float16* pooled = (_Float16*)carve((size_t)12845056 * 2);  // 1024x12544
  _Float16* w1h    = (_Float16*)carve((size_t)12845056 * 2);
  _Float16* w2h    = (_Float16*)carve((size_t)1048576 * 2);
  _Float16* h1h    = (_Float16*)carve((size_t)1048576 * 2);
  float* h2f       = (float*)carve((size_t)1048576 * 4);
  float* cls_out   = (float*)carve((size_t)(1024 * 91) * 4);
  float* reg_out   = (float*)carve((size_t)(1024 * 364) * 4);
  float4* cb       = (float4*)carve((size_t)92160 * 16);
  float* cs        = (float*)carve((size_t)92160 * 4);
  float4* kb       = (float4*)carve((size_t)18000 * 16);
  float* sc2       = (float*)carve((size_t)18000 * 4);

  auto nb = [](int n) { return (n + 255) / 256; };

  int n;
  n = 2217984; k_normalize<<<nb(n), 256, 0, stream>>>(images, xn, n);
  n = 2957312; k_conv<<<nb(n), 256, 0, stream>>>(xn, bb_w1, bb_b1, f1, 3, 608, 608, 64, 152, 152, 7, 4, 3, 1, n);
  n = 1478656; k_conv<<<nb(n), 256, 0, stream>>>(f1, bb_w2, bb_b2, f2, 64, 152, 152, 128, 76, 76, 3, 2, 1, 1, n);
  n = 739328;  k_conv<<<nb(n), 256, 0, stream>>>(f2, bb_w3, bb_b3, f3, 128, 76, 76, 256, 38, 38, 3, 2, 1, 1, n);
  n = 739328;  k_conv<<<nb(n), 256, 0, stream>>>(f3, rpn_cw, rpn_cb, trpn, 256, 38, 38, 256, 38, 38, 3, 1, 1, 1, n);
  n = 25992;   k_conv<<<nb(n), 256, 0, stream>>>(trpn, rpn_sw, rpn_sb, objb, 256, 38, 38, 9, 38, 38, 1, 1, 0, 0, n);
  n = 103968;  k_conv<<<nb(n), 256, 0, stream>>>(trpn, rpn_rw, rpn_rb, regb, 256, 38, 38, 36, 38, 38, 1, 1, 0, 0, n);
  n = 25992;   k_props<<<nb(n), 256, 0, stream>>>(objb, regb, props, scores, n);
  k_topk_pre<<<2, 256, 0, stream>>>(scores, (const float4*)props, top_s, top_b);
  k_rpn_nms<<<2, 256, 0, stream>>>(top_s, top_b, pboxes, validb);
  n = 12845056; k_roi_align<<<nb(n), 256, 0, stream>>>(f3, pboxes, pooled, n);
  n = 12845056; k_cvt<<<nb(n), 256, 0, stream>>>(fc1_w, w1h, n);
  n = 1048576;  k_cvt<<<nb(n), 256, 0, stream>>>(fc2_w, w2h, n);
  {
    dim3 g1(1024 / NBLK, 1024 / MBLK);
    k_gemm_wmma<<<g1, 256, 0, stream>>>(pooled, w1h, fc1_b, (float*)nullptr, h1h, 1024, 1024, 12544, 1);
    k_gemm_wmma<<<g1, 256, 0, stream>>>(h1h, w2h, fc2_b, h2f, (_Float16*)nullptr, 1024, 1024, 1024, 1);
  }
  n = 1024 * 91;  k_fc<<<nb(n), 256, 0, stream>>>(h2f, cls_w, cls_b, cls_out, 1024, 91, n);
  n = 1024 * 364; k_fc<<<nb(n), 256, 0, stream>>>(h2f, reg_w, reg_b, reg_out, 1024, 364, n);
  n = 1024; k_post<<<nb(n), 256, 0, stream>>>(cls_out, reg_out, pboxes, validb, cb, cs, n);
  k_class_nms<<<180, 256, 0, stream>>>(cb, cs, kb, sc2);
  k_final<<<2, 256, 0, stream>>>(sc2, kb, (float*)d_out);
}
